// LCF_ATEPC_23398981828955
// MI455X (gfx1250) — matrix operand order repaired and checked by reading, not compiled, not run on hardware
//
#include <hip/hip_runtime.h>
#include <hip/hip_bf16.h>
#include <math.h>

// ---------------------------------------------------------------------------
// LCF-ATEPC head for MI455X (gfx1250, wave32).
// All GEMMs: v_wmma_f32_16x16x32_bf16 (bf16 A/B, f32 accumulate).
// Weights pre-converted to bf16 once; fragments load as b128.
// M-blocking: 2 row-tiles per block share every B fragment (halves L2 weight
// traffic, the dominant byte stream at 23.3 TB/s HBM / ~100MB activations).
// ---------------------------------------------------------------------------

typedef float v8f_t __attribute__((ext_vector_type(8)));
typedef __bf16 v16bf_t __attribute__((ext_vector_type(16)));

union Frag {
  v16bf_t v;
  unsigned int u[8];
  uint4 q[2];
};

static constexpr int kB  = 64;
static constexpr int kS  = 512;
static constexpr int kH  = 768;
static constexpr int kH2 = 384;
static constexpr int kNS = 4;
static constexpr int kCW = 3;
static constexpr int kW  = 15;
static constexpr int kSpans = kB * kNS;        // 256
static constexpr int kAteOut = kB * kS * 2;    // 65536

__device__ __forceinline__ unsigned short f2bf(float f) {
  __bf16 h = (__bf16)f;                       // hw v_cvt, RNE
  return __builtin_bit_cast(unsigned short, h);
}
__device__ __forceinline__ float bf2f(unsigned short h) {
  return __uint_as_float(((unsigned int)h) << 16);
}
__device__ __forceinline__ unsigned int pack2(float a, float b) {
  union { __bf16 h[2]; unsigned int u; } t;
  t.h[0] = (__bf16)a; t.h[1] = (__bf16)b;
  return t.u;
}
__device__ __forceinline__ float gelu_exact(float x) {
  return 0.5f * x * (1.0f + erff(x * 0.7071067811865476f));
}
__device__ __forceinline__ v8f_t vzero() {
  v8f_t z;
#pragma unroll
  for (int i = 0; i < 8; ++i) z[i] = 0.0f;
  return z;
}
__device__ __forceinline__ v8f_t wmma_bf16(Frag a, Frag b, v8f_t c) {
  return __builtin_amdgcn_wmma_f32_16x16x32_bf16(false, a.v, false, b.v,
                                                 (short)0, c, false, false);
}
// A fragment (16x32 bf16) from LDS; row = lane&15, 16 consecutive K values
// (k + 0/16 per lane half). Two 16B ds loads (pairs 0..3 and 4..7 relative to base).
__device__ __forceinline__ Frag load_afrag(const unsigned int* A, int rowStrideU,
                                           int lane, int k) {
  int lm = lane & 15;
  int khalf = (lane < 16) ? 0 : 16;
  const unsigned int* p = A + lm * rowStrideU + ((k + khalf) >> 1);
  Frag a;
  a.q[0] = *(const uint4*)p;
  a.q[1] = *(const uint4*)(p + 4);
  return a;
}
// B fragment (32x16 bf16): lane holds column n, 16 consecutive K values
// (k + 0/16 per lane half) from pre-converted bf16 weights. Two b128 loads.
__device__ __forceinline__ Frag load_bfrag(const unsigned int* wb) {
  Frag b;
  const uint4* p = (const uint4*)wb;
  b.q[0] = p[0];
  b.q[1] = p[1];
  return b;
}

// ---------------------------------------------------------------------------
// K0: f32 -> packed bf16 weight conversion (run once per launch, up front).
// ---------------------------------------------------------------------------
__global__ __launch_bounds__(256) void k_cvt(const float* __restrict__ src,
                                             unsigned int* __restrict__ dst,
                                             int npairs) {
  const float2* s2 = (const float2*)src;
  int i = blockIdx.x * 256 + threadIdx.x;
  int stride = gridDim.x * 256;
  for (; i < npairs; i += stride) {
    float2 v = s2[i];
    dst[i] = pack2(v.x, v.y);
  }
}

// ---------------------------------------------------------------------------
// K1: ATE head. One block = 32 rows of [B*S]. LN -> GEMM(768->384, WMMA,
//     2 row-tiles x 3 col-tiles per wave) -> GELU -> LN -> 384->2 -> logits.
// ---------------------------------------------------------------------------
__global__ __launch_bounds__(256) void k_ate(
    const float* __restrict__ seq,
    const float* __restrict__ g0, const float* __restrict__ b0,
    const unsigned int* __restrict__ wb1, const float* __restrict__ b1,
    const float* __restrict__ lng, const float* __restrict__ lnb,
    const float* __restrict__ w2, const float* __restrict__ b2,
    float* __restrict__ out) {
  __shared__ __align__(16) char smem[32 * kH2 * 4];  // 48KB: bf16 A, then f32 H1
  __shared__ float red[256], red2[256];
  __shared__ float sMean[32], sRstd[32];
  unsigned int* A = (unsigned int*)smem;
  float* H1 = (float*)smem;

  int tid = threadIdx.x, lane = tid & 31, wid = tid >> 5;
  int r = tid >> 3, c = tid & 7;            // 8 threads per row, 32 rows
  size_t rowbase = (size_t)blockIdx.x * 32;

  // input LayerNorm stats (per row)
  const float* rp = seq + (rowbase + r) * kH;
  float s = 0.f, s2 = 0.f;
  for (int h = c; h < kH; h += 8) { float x = rp[h]; s += x; s2 += x * x; }
  red[tid] = s; red2[tid] = s2;
  __syncthreads();
  if (c == 0) {
    float a = 0.f, q = 0.f;
    for (int i = 0; i < 8; ++i) { a += red[r * 8 + i]; q += red2[r * 8 + i]; }
    float m = a / kH;
    sMean[r] = m; sRstd[r] = rsqrtf(q / kH - m * m + 1e-5f);
  }
  __syncthreads();
  // normalized bf16 A tile into LDS (32 x 768 bf16)
  for (int p = tid; p < 32 * (kH / 2); p += 256) {
    int rr = p / (kH / 2); int h = (p % (kH / 2)) * 2;
    const float* q = seq + (rowbase + rr) * kH + h;
    float m = sMean[rr], rs = sRstd[rr];
    float x0 = (q[0] - m) * rs * g0[h] + b0[h];
    float x1 = (q[1] - m) * rs * g0[h + 1] + b0[h + 1];
    A[p] = pack2(x0, x1);
  }
  __syncthreads();

  // GEMM1: [32x768] @ w1^T -> [32x384]; 3 col-tiles x 2 row-tiles per wave
  int lm = lane & 15;
  int bkh = (lane < 16) ? 0 : 16;
  int mbase = (lane < 16) ? 0 : 8;
  int n0 = wid * 48;
  v8f_t acc[2][3];
#pragma unroll
  for (int rt = 0; rt < 2; ++rt)
#pragma unroll
    for (int t = 0; t < 3; ++t) acc[rt][t] = vzero();
  for (int k = 0; k < kH; k += 32) {
    Frag a0 = load_afrag(A, kH / 2, lane, k);
    Frag a1 = load_afrag(A + 16 * (kH / 2), kH / 2, lane, k);
#pragma unroll
    for (int t = 0; t < 3; ++t) {
      int n = n0 + t * 16 + lm;
      Frag b = load_bfrag(wb1 + (size_t)n * (kH / 2) + ((k + bkh) >> 1));
      acc[0][t] = wmma_bf16(a0, b, acc[0][t]);
      acc[1][t] = wmma_bf16(a1, b, acc[1][t]);
    }
  }
  __syncthreads();   // done reading A; reuse LDS as f32 H1 (32 x 384)
#pragma unroll
  for (int rt = 0; rt < 2; ++rt)
#pragma unroll
    for (int t = 0; t < 3; ++t) {
      int n = n0 + t * 16 + lm;
      float bias = b1[n];
#pragma unroll
      for (int rr = 0; rr < 8; ++rr)
        H1[(rt * 16 + mbase + rr) * kH2 + n] = gelu_exact(acc[rt][t][rr] + bias);
    }
  __syncthreads();

  // LayerNorm over H2
  s = 0.f; s2 = 0.f;
  for (int h = c; h < kH2; h += 8) { float x = H1[r * kH2 + h]; s += x; s2 += x * x; }
  red[tid] = s; red2[tid] = s2;
  __syncthreads();
  if (c == 0) {
    float a = 0.f, q = 0.f;
    for (int i = 0; i < 8; ++i) { a += red[r * 8 + i]; q += red2[r * 8 + i]; }
    float m = a / kH2;
    sMean[r] = m; sRstd[r] = rsqrtf(q / kH2 - m * m + 1e-5f);
  }
  __syncthreads();
  // final 384 -> 2 projection
  float m = sMean[r], rs = sRstd[r];
  float p0 = 0.f, p1 = 0.f;
  for (int h = c; h < kH2; h += 8) {
    float xn = (H1[r * kH2 + h] - m) * rs * lng[h] + lnb[h];
    p0 += xn * w2[h]; p1 += xn * w2[kH2 + h];
  }
  red[tid] = p0; red2[tid] = p1;
  __syncthreads();
  if (c == 0) {
    float a = 0.f, q = 0.f;
    for (int i = 0; i < 8; ++i) { a += red[r * 8 + i]; q += red2[r * 8 + i]; }
    size_t grow = rowbase + r;
    out[grow * 2 + 0] = a + b2[0];
    out[grow * 2 + 1] = q + b2[1];
  }
}

// ---------------------------------------------------------------------------
// K2a: per-span gather: aspect mean + zero-padded bf16 context tile [16x768].
// ---------------------------------------------------------------------------
__global__ __launch_bounds__(256) void k_prep(
    const float* __restrict__ seq, const int* __restrict__ st,
    const int* __restrict__ en, float* __restrict__ aspmean,
    unsigned int* __restrict__ ctxbf, int* __restrict__ nvalids) {
  int sp = blockIdx.x; int b = sp >> 2;
  int tid = threadIdx.x;
  int s = st[sp]; s = (s < 0) ? 0 : ((s > kS - 1) ? kS - 1 : s);
  int e = en[sp]; e = (e > kS - 1) ? kS - 1 : e; if (e < s) e = s;
  int left = s - kCW; if (left < 0) left = 0;
  int right = e + 1 + kCW; if (right > kS) right = kS;
  int nval = right - left;
  int cnt = e - s;
  const float* sb = seq + (size_t)b * kS * kH;
  for (int h = tid; h < kH; h += 256) {
    float acc = 0.f;
    for (int t = s; t < e; ++t) acc += sb[(size_t)t * kH + h];
    aspmean[(size_t)sp * kH + h] = (cnt > 0) ? (acc / (float)cnt) : sb[h];
  }
  for (int p = tid; p < 16 * (kH / 2); p += 256) {
    int w = p / (kH / 2); int h = (p % (kH / 2)) * 2;
    unsigned int val = 0u;
    if (w < nval) {
      int tok = left + w; if (tok > kS - 1) tok = kS - 1;
      const float* q = sb + (size_t)tok * kH + h;
      val = pack2(q[0], q[1]);
    }
    ctxbf[(size_t)sp * (16 * (kH / 2)) + p] = val;
  }
  if (tid == 0) nvalids[sp] = nval;
}

// ---------------------------------------------------------------------------
// K2b: QKV for 2 spans/block: [2x16x768] @ w_in^T -> [2x16x2304] bf16.
//      Each B fragment feeds both span tiles; 6 groups of 3 col-tiles/wave.
// ---------------------------------------------------------------------------
__global__ __launch_bounds__(256) void k_qkv(
    const unsigned int* __restrict__ ctxbf, const unsigned int* __restrict__ wbin,
    const float* __restrict__ b_in, unsigned short* __restrict__ qkvbf) {
  __shared__ __align__(16) unsigned int sA[2 * 16 * (kH / 2)];   // 48KB
  int sp0 = blockIdx.x * 2;
  int tid = threadIdx.x, lane = tid & 31, wid = tid >> 5;
  for (int p = tid; p < 2 * 16 * (kH / 2); p += 256)
    sA[p] = ctxbf[(size_t)sp0 * (16 * (kH / 2)) + p];
  __syncthreads();
  int lm = lane & 15, bkh = (lane < 16) ? 0 : 16, mbase = (lane < 16) ? 0 : 8;
  for (int g = 0; g < 6; ++g) {
    int nbase = (wid * 18 + g * 3) * 16;
    v8f_t acc[2][3];
#pragma unroll
    for (int spn = 0; spn < 2; ++spn)
#pragma unroll
      for (int t = 0; t < 3; ++t) acc[spn][t] = vzero();
    for (int k = 0; k < kH; k += 32) {
      Frag a0 = load_afrag(sA, kH / 2, lane, k);
      Frag a1 = load_afrag(sA + 16 * (kH / 2), kH / 2, lane, k);
#pragma unroll
      for (int t = 0; t < 3; ++t) {
        int n = nbase + t * 16 + lm;
        Frag b = load_bfrag(wbin + (size_t)n * (kH / 2) + ((k + bkh) >> 1));
        acc[0][t] = wmma_bf16(a0, b, acc[0][t]);
        acc[1][t] = wmma_bf16(a1, b, acc[1][t]);
      }
    }
#pragma unroll
    for (int spn = 0; spn < 2; ++spn) {
      unsigned short* outp = qkvbf + (size_t)(sp0 + spn) * 16 * 2304;
#pragma unroll
      for (int t = 0; t < 3; ++t) {
        int n = nbase + t * 16 + lm;
        float bias = b_in[n];
#pragma unroll
        for (int rr = 0; rr < 8; ++rr)
          outp[(size_t)(mbase + rr) * 2304 + n] = f2bf(acc[spn][t][rr] + bias);
      }
    }
  }
}

// ---------------------------------------------------------------------------
// K2c: tiny 8-head 15x15 attention per span (VALU); writes bf16 attn-out.
// ---------------------------------------------------------------------------
__global__ __launch_bounds__(256) void k_attn(
    const unsigned short* __restrict__ qkvbf, const int* __restrict__ nvalids,
    unsigned short* __restrict__ attnbf) {
  int sp = blockIdx.x; int tid = threadIdx.x;
  const unsigned short* base = qkvbf + (size_t)sp * 16 * 2304;
  unsigned short* ob = attnbf + (size_t)sp * 16 * kH;
  int nval = nvalids[sp];
  if (tid < 120) {
    int head = tid / kW, qi = tid % kW;
    const unsigned short* qp = base + (size_t)qi * 2304 + head * 96;
    float qreg[96];
#pragma unroll 8
    for (int t = 0; t < 96; ++t) qreg[t] = bf2f(qp[t]);
    float sc[kW];
    float mx = -3.0e38f;
    for (int kk = 0; kk < kW; ++kk) {
      float d = -3.0e38f;
      if (kk < nval) {
        const unsigned short* kp = base + (size_t)kk * 2304 + 768 + head * 96;
        float acc = 0.f;
        for (int t = 0; t < 96; ++t) acc += qreg[t] * bf2f(kp[t]);
        d = acc * 0.10206207261596577f;   // 96^-0.5
      }
      sc[kk] = d; if (d > mx) mx = d;
    }
    float den = 0.f;
    for (int kk = 0; kk < kW; ++kk) {
      float e2 = (sc[kk] > -1.0e38f) ? expf(sc[kk] - mx) : 0.f;
      sc[kk] = e2; den += e2;
    }
    float inv = 1.f / den;
    for (int d0 = 0; d0 < 96; ++d0) {
      float o = 0.f;
      for (int kk = 0; kk < kW; ++kk)
        o += sc[kk] * bf2f(base[(size_t)kk * 2304 + 1536 + head * 96 + d0]);
      ob[(size_t)qi * kH + head * 96 + d0] = f2bf(o * inv);
    }
  } else {
    for (int idx = tid - 120; idx < kH; idx += 136)  // zero pad row 15
      ob[(size_t)15 * kH + idx] = 0;
  }
}

// ---------------------------------------------------------------------------
// K2d: out-projection for 2 spans/block (WMMA) + bias + masked row mean.
// ---------------------------------------------------------------------------
__global__ __launch_bounds__(256) void k_outproj(
    const unsigned short* __restrict__ attnbf, const unsigned int* __restrict__ wbout,
    const float* __restrict__ b_out, const int* __restrict__ nvalids,
    float* __restrict__ ctxmean) {
  __shared__ __align__(16) unsigned int sA[2 * 16 * (kH / 2)];   // 48KB
  int sp0 = blockIdx.x * 2;
  int tid = threadIdx.x, lane = tid & 31, wid = tid >> 5;
  const unsigned int* src = (const unsigned int*)attnbf + (size_t)sp0 * (16 * (kH / 2));
  for (int p = tid; p < 2 * 16 * (kH / 2); p += 256) sA[p] = src[p];
  __syncthreads();
  int nv0 = nvalids[sp0], nv1 = nvalids[sp0 + 1];
  int lm = lane & 15, bkh = (lane < 16) ? 0 : 16, mbase = (lane < 16) ? 0 : 8;
  for (int g = 0; g < 2; ++g) {
    int nbase = (wid * 6 + g * 3) * 16;
    v8f_t acc[2][3];
#pragma unroll
    for (int spn = 0; spn < 2; ++spn)
#pragma unroll
      for (int t = 0; t < 3; ++t) acc[spn][t] = vzero();
    for (int k = 0; k < kH; k += 32) {
      Frag a0 = load_afrag(sA, kH / 2, lane, k);
      Frag a1 = load_afrag(sA + 16 * (kH / 2), kH / 2, lane, k);
#pragma unroll
      for (int t = 0; t < 3; ++t) {
        int n = nbase + t * 16 + lm;
        Frag b = load_bfrag(wbout + (size_t)n * (kH / 2) + ((k + bkh) >> 1));
        acc[0][t] = wmma_bf16(a0, b, acc[0][t]);
        acc[1][t] = wmma_bf16(a1, b, acc[1][t]);
      }
    }
#pragma unroll
    for (int spn = 0; spn < 2; ++spn) {
      int nv = spn ? nv1 : nv0;
      float invn = 1.f / (float)nv;
#pragma unroll
      for (int t = 0; t < 3; ++t) {
        int n = nbase + t * 16 + lm;
        float part = 0.f;
#pragma unroll
        for (int rr = 0; rr < 8; ++rr)
          if (mbase + rr < nv) part += acc[spn][t][rr];
        float tot = part + __shfl_xor(part, 16, 32);  // combine row halves
        if (lane < 16)
          ctxmean[(size_t)(sp0 + spn) * kH + n] = tot * invn + b_out[n];
      }
    }
  }
}

// ---------------------------------------------------------------------------
// K2e: fusion GEMM: [256x1536] (concat asp|ctx) @ fus_w^T -> GELU -> [256x768]
// ---------------------------------------------------------------------------
__global__ __launch_bounds__(256) void k_fuse(
    const float* __restrict__ aspmean, const float* __restrict__ ctxmean,
    const unsigned int* __restrict__ wbfw, const float* __restrict__ fb,
    float* __restrict__ fused) {
  __shared__ __align__(16) unsigned int sA[16 * 768];   // 16 x 1536 bf16 = 48KB
  int tid = threadIdx.x, lane = tid & 31, wid = tid >> 5;
  int row0 = blockIdx.x * 16;
  for (int p = tid; p < 16 * 768; p += 256) {
    int rr = p / 768; int h = (p % 768) * 2;
    int span = row0 + rr;
    float x0, x1;
    if (h < kH) { x0 = aspmean[(size_t)span * kH + h]; x1 = aspmean[(size_t)span * kH + h + 1]; }
    else        { x0 = ctxmean[(size_t)span * kH + h - kH]; x1 = ctxmean[(size_t)span * kH + h - kH + 1]; }
    sA[p] = pack2(x0, x1);
  }
  __syncthreads();
  int lm = lane & 15, bkh = (lane < 16) ? 0 : 16, mbase = (lane < 16) ? 0 : 8;
  v8f_t acc[6];
#pragma unroll
  for (int t = 0; t < 6; ++t) acc[t] = vzero();
  int nbase = wid * 96;
  for (int k = 0; k < 1536; k += 32) {
    Frag a = load_afrag(sA, 768, lane, k);
#pragma unroll
    for (int t = 0; t < 6; ++t) {
      int n = nbase + t * 16 + lm;
      Frag b = load_bfrag(wbfw + (size_t)n * 768 + ((k + bkh) >> 1));
      acc[t] = wmma_bf16(a, b, acc[t]);
    }
  }
#pragma unroll
  for (int t = 0; t < 6; ++t) {
    int n = nbase + t * 16 + lm;
    float bias = fb[n];
#pragma unroll
    for (int rr = 0; rr < 8; ++rr) {
      int span = row0 + mbase + rr;
      fused[(size_t)span * kH + n] = gelu_exact(acc[t][rr] + bias);
    }
  }
}

// ---------------------------------------------------------------------------
// K2f: APC classifier: span-mean -> LN -> 768->384 GELU -> LN -> 384->2.
// ---------------------------------------------------------------------------
__global__ __launch_bounds__(256) void k_apc(
    const float* __restrict__ fused,
    const float* __restrict__ apc_g, const float* __restrict__ apc_b,
    const float* __restrict__ w1, const float* __restrict__ b1,
    const float* __restrict__ lng, const float* __restrict__ lnb,
    const float* __restrict__ w2, const float* __restrict__ b2,
    float* __restrict__ out) {
  __shared__ float feat[kH];
  __shared__ float h1[kH2];
  __shared__ float red[256], red2[256];
  __shared__ float sM, sR;
  int b = blockIdx.x; int tid = threadIdx.x;
  const float* fp = fused + (size_t)b * kNS * kH;
  float s = 0.f, s2 = 0.f;
  for (int h = tid; h < kH; h += 256) {
    float v = 0.25f * (fp[h] + fp[kH + h] + fp[2 * kH + h] + fp[3 * kH + h]);
    feat[h] = v; s += v; s2 += v * v;
  }
  red[tid] = s; red2[tid] = s2; __syncthreads();
  if (tid == 0) {
    float a = 0.f, q = 0.f;
    for (int i = 0; i < 256; ++i) { a += red[i]; q += red2[i]; }
    float m = a / kH; sM = m; sR = rsqrtf(q / kH - m * m + 1e-5f);
  }
  __syncthreads();
  float m = sM, rs = sR;
  for (int h = tid; h < kH; h += 256)
    feat[h] = (feat[h] - m) * rs * apc_g[h] + apc_b[h];
  __syncthreads();
  for (int j = tid; j < kH2; j += 256) {
    float acc = 0.f;
    const float* wr = w1 + (size_t)j * kH;
    for (int h = 0; h < kH; ++h) acc += feat[h] * wr[h];
    h1[j] = gelu_exact(acc + b1[j]);
  }
  __syncthreads();
  s = 0.f; s2 = 0.f;
  for (int h = tid; h < kH2; h += 256) { float v = h1[h]; s += v; s2 += v * v; }
  red[tid] = s; red2[tid] = s2; __syncthreads();
  if (tid == 0) {
    float a = 0.f, q = 0.f;
    for (int i = 0; i < 256; ++i) { a += red[i]; q += red2[i]; }
    float mm = a / kH2; sM = mm; sR = rsqrtf(q / kH2 - mm * mm + 1e-5f);
  }
  __syncthreads();
  m = sM; rs = sR;
  float p0 = 0.f, p1 = 0.f;
  for (int h = tid; h < kH2; h += 256) {
    float xn = (h1[h] - m) * rs * lng[h] + lnb[h];
    p0 += xn * w2[h]; p1 += xn * w2[kH2 + h];
  }
  __syncthreads();
  red[tid] = p0; red2[tid] = p1; __syncthreads();
  if (tid == 0) {
    float a = 0.f, q = 0.f;
    for (int i = 0; i < 256; ++i) { a += red[i]; q += red2[i]; }
    out[kAteOut + b * 2 + 0] = a + b2[0];
    out[kAteOut + b * 2 + 1] = q + b2[1];
  }
}

// ---------------------------------------------------------------------------
extern "C" void kernel_launch(void* const* d_in, const int* in_sizes, int n_in,
                              void* d_out, int out_size, void* d_ws, size_t ws_size,
                              hipStream_t stream) {
  (void)in_sizes; (void)n_in; (void)out_size; (void)ws_size;
  const float* seq    = (const float*)d_in[0];
  const int*   st     = (const int*)d_in[1];
  const int*   en     = (const int*)d_in[2];
  const float* ate_g  = (const float*)d_in[3];
  const float* ate_b  = (const float*)d_in[4];
  const float* apc_g  = (const float*)d_in[5];
  const float* apc_b  = (const float*)d_in[6];
  const float* w_in   = (const float*)d_in[7];
  const float* b_in   = (const float*)d_in[8];
  const float* w_out  = (const float*)d_in[9];
  const float* b_out  = (const float*)d_in[10];
  const float* fus_w  = (const float*)d_in[11];
  const float* fus_b  = (const float*)d_in[12];
  const float* asp_w1 = (const float*)d_in[13];
  const float* asp_b1 = (const float*)d_in[14];
  const float* asp_lng= (const float*)d_in[15];
  const float* asp_lnb= (const float*)d_in[16];
  const float* asp_w2 = (const float*)d_in[17];
  const float* asp_b2 = (const float*)d_in[18];
  const float* sen_w1 = (const float*)d_in[19];
  const float* sen_b1 = (const float*)d_in[20];
  const float* sen_lng= (const float*)d_in[21];
  const float* sen_lnb= (const float*)d_in[22];
  const float* sen_w2 = (const float*)d_in[23];
  const float* sen_b2 = (const float*)d_in[24];
  float* out = (float*)d_out;

  char* ws = (char*)d_ws;
  size_t o_ctx   = 0;                                       // [256][16][768] bf16
  size_t o_qkv   = o_ctx  + (size_t)kSpans * 16 * kH * 2;   //  6,291,456
  size_t o_attn  = o_qkv  + (size_t)kSpans * 16 * 2304 * 2; // 25,165,824
  size_t o_asp   = o_attn + (size_t)kSpans * 16 * kH * 2;   // 31,457,280
  size_t o_cm    = o_asp  + (size_t)kSpans * kH * 4;        // 32,243,712
  size_t o_fu    = o_cm   + (size_t)kSpans * kH * 4;        // 33,030,144
  size_t o_nv    = o_fu   + (size_t)kSpans * kH * 4;        // 33,816,576
  size_t o_w1b   = o_nv   + 1024;                           // 33,817,600 (256-aligned)
  size_t o_winb  = o_w1b  + (size_t)kH2 * kH * 2;           // 34,407,424
  size_t o_woutb = o_winb + (size_t)3 * kH * kH * 2;        // 37,946,368
  size_t o_wfwb  = o_woutb + (size_t)kH * kH * 2;           // 39,126,016

  unsigned int*   ctxbf   = (unsigned int*)(ws + o_ctx);
  unsigned short* qkvbf   = (unsigned short*)(ws + o_qkv);
  unsigned short* attnbf  = (unsigned short*)(ws + o_attn);
  float*          aspmean = (float*)(ws + o_asp);
  float*          ctxmean = (float*)(ws + o_cm);
  float*          fusedp  = (float*)(ws + o_fu);
  int*            nvalids = (int*)(ws + o_nv);
  unsigned int*   wb1     = (unsigned int*)(ws + o_w1b);
  unsigned int*   wbin    = (unsigned int*)(ws + o_winb);
  unsigned int*   wbout   = (unsigned int*)(ws + o_woutb);
  unsigned int*   wbfw    = (unsigned int*)(ws + o_wfwb);

  // weight pre-conversion (f32 -> packed bf16)
  k_cvt<<<288, 256, 0, stream>>>(asp_w1, wb1,  kH2 * kH / 2);       // 147456
  k_cvt<<<512, 256, 0, stream>>>(w_in,   wbin, 3 * kH * kH / 2);    // 884736
  k_cvt<<<512, 256, 0, stream>>>(w_out,  wbout, kH * kH / 2);       // 294912
  k_cvt<<<512, 256, 0, stream>>>(fus_w,  wbfw, kH * 1536 / 2);      // 589824

  k_ate<<<(kB * kS) / 32, 256, 0, stream>>>(seq, ate_g, ate_b,
      wb1, asp_b1, asp_lng, asp_lnb, asp_w2, asp_b2, out);
  k_prep<<<kSpans, 256, 0, stream>>>(seq, st, en, aspmean, ctxbf, nvalids);
  k_qkv<<<kSpans / 2, 256, 0, stream>>>(ctxbf, wbin, b_in, qkvbf);
  k_attn<<<kSpans, 256, 0, stream>>>(qkvbf, nvalids, attnbf);
  k_outproj<<<kSpans / 2, 256, 0, stream>>>(attnbf, wbout, b_out, nvalids, ctxmean);
  k_fuse<<<kSpans / 16, 256, 0, stream>>>(aspmean, ctxmean, wbfw, fus_b, fusedp);
  k_apc<<<kB, 256, 0, stream>>>(fusedp, apc_g, apc_b,
      sen_w1, sen_b1, sen_lng, sen_lnb, sen_w2, sen_b2, out);
}
